// GrigSearchGNNEncoder_64707977281830
// MI455X (gfx1250) — compile-verified, matrix operand order checked by reading, not run on hardware
//
#include <hip/hip_runtime.h>
#include <hip/hip_bf16.h>

typedef __attribute__((ext_vector_type(16))) _Float16 v16h;
typedef __attribute__((ext_vector_type(8)))  float    v8f;

#define WAVE 32

// ---------------------------------------------------------------------------
// K-index interleave for a 16-bit 16x32 A/B WMMA fragment (ISA 7.12.2):
// VGPR v (v = e>>1), lane-half h: K = 16*(v>>2) + 8*h + 2*(v&3) + (e&1)
// Fragment loader: 'row' points at 32 consecutive f32 K-values for this
// lane's matrix row (A: node row; B: weight row n of W, i.e. column n of W^T).
// ---------------------------------------------------------------------------
__device__ __forceinline__ v16h load_frag32(const float* __restrict__ row, int half) {
    v16h f;
#pragma unroll
    for (int v = 0; v < 8; ++v) {
        const int k = ((v >> 2) << 4) + (half << 3) + ((v & 3) << 1);
        const float2 t = *reinterpret_cast<const float2*>(row + k);
        f[2 * v]     = (_Float16)t.x;
        f[2 * v + 1] = (_Float16)t.y;
    }
    return f;
}

// ---------------------------------------------------------------------------
// Zero a float4 region (ws is poisoned 0xAA; atomics need clean accumulators
// on every call, including graph replays).
// ---------------------------------------------------------------------------
__global__ void zero4_kernel(float4* __restrict__ p, long long n4) {
    long long t = (long long)blockIdx.x * blockDim.x + threadIdx.x;
    if (t < n4) p[t] = make_float4(0.f, 0.f, 0.f, 0.f);
}

// ---------------------------------------------------------------------------
// Edge scatter: agg[dst] += feat[src].  One thread per (edge, float4 chunk).
// chunkShift = log2(d/4).  Random 16B gathers + fp32 no-return hardware
// atomics; all tables are L2-resident (<=25.6MB each vs 192MB L2).
// ---------------------------------------------------------------------------
__global__ void scatter_kernel(const int* __restrict__ src,
                               const int* __restrict__ dst,
                               const float* __restrict__ feat,
                               float* __restrict__ agg,
                               int nEdges, int chunkShift) {
    long long t = (long long)blockIdx.x * blockDim.x + threadIdx.x;
    int e = (int)(t >> chunkShift);
    if (e >= nEdges) return;
    const int c = ((int)t & ((1 << chunkShift) - 1)) << 2;   // float offset
    const int d = 4 << chunkShift;                            // 32 or 64
    const int s  = src[e];
    const int dd = dst[e];
    const float4 v = *reinterpret_cast<const float4*>(feat + (long long)s * d + c);
    float* p = agg + (long long)dd * d + c;
    unsafeAtomicAdd(p + 0, v.x);
    unsafeAtomicAdd(p + 1, v.y);
    unsafeAtomicAdd(p + 2, v.z);
    unsafeAtomicAdd(p + 3, v.w);
}

// ---------------------------------------------------------------------------
// Layer 1: h[16x64 tile] = relu( agg1[16x32] @ Wl^T + b + x[16x32] @ Wr^T )
// One wave per 16-row tile; 4 N-tiles of 16x16; K=32 in one WMMA each.
// ---------------------------------------------------------------------------
__global__ void __launch_bounds__(256)
gemm1_kernel(const float* __restrict__ agg, const float* __restrict__ x,
             const float* __restrict__ Wl,  const float* __restrict__ bl,
             const float* __restrict__ Wr,  float* __restrict__ h, int nTiles) {
    const int wave = blockIdx.x * (blockDim.x >> 5) + (threadIdx.x >> 5);
    if (wave >= nTiles) return;                 // wave-uniform: EXEC all-1s below
    const int lane = threadIdx.x & 31;
    const int half = lane >> 4;
    const int idx  = lane & 15;                 // A: row m; B: col n; C/D: col n
    const int m0   = wave * 16;

    const v16h a_agg = load_frag32(agg + (size_t)(m0 + idx) * 32, half);
    const v16h a_x   = load_frag32(x   + (size_t)(m0 + idx) * 32, half);

#pragma unroll
    for (int nt = 0; nt < 4; ++nt) {
        const int n0 = nt * 16;
        const v16h b_l = load_frag32(Wl + (size_t)(n0 + idx) * 32, half);
        const v16h b_r = load_frag32(Wr + (size_t)(n0 + idx) * 32, half);
        v8f c = {};
        c = __builtin_amdgcn_wmma_f32_16x16x32_f16(false, a_agg, false, b_l,
                                                   (short)0, c, false, false);
        c = __builtin_amdgcn_wmma_f32_16x16x32_f16(false, a_x,   false, b_r,
                                                   (short)0, c, false, false);
        const float bias = bl[n0 + idx];
#pragma unroll
        for (int r = 0; r < 8; ++r) {
            float val = c[r] + bias;
            val = val > 0.f ? val : 0.f;        // relu
            h[(size_t)(m0 + r + 8 * half) * 64 + n0 + idx] = val;
        }
    }
}

// ---------------------------------------------------------------------------
// Layer 2: out[16x32 tile] = agg2[16x64] @ Wl^T + b + h[16x64] @ Wr^T
// One wave per 16-row tile; 2 N-tiles; K=64 split into 2 WMMA K-chunks.
// ---------------------------------------------------------------------------
__global__ void __launch_bounds__(256)
gemm2_kernel(const float* __restrict__ agg, const float* __restrict__ h,
             const float* __restrict__ Wl,  const float* __restrict__ bl,
             const float* __restrict__ Wr,  float* __restrict__ out, int nTiles) {
    const int wave = blockIdx.x * (blockDim.x >> 5) + (threadIdx.x >> 5);
    if (wave >= nTiles) return;
    const int lane = threadIdx.x & 31;
    const int half = lane >> 4;
    const int idx  = lane & 15;
    const int m0   = wave * 16;

    v16h a_a[2], a_h[2];
#pragma unroll
    for (int kc = 0; kc < 2; ++kc) {
        a_a[kc] = load_frag32(agg + (size_t)(m0 + idx) * 64 + 32 * kc, half);
        a_h[kc] = load_frag32(h   + (size_t)(m0 + idx) * 64 + 32 * kc, half);
    }

#pragma unroll
    for (int nt = 0; nt < 2; ++nt) {
        const int n0 = nt * 16;
        v8f c = {};
#pragma unroll
        for (int kc = 0; kc < 2; ++kc) {
            const v16h b_l = load_frag32(Wl + (size_t)(n0 + idx) * 64 + 32 * kc, half);
            c = __builtin_amdgcn_wmma_f32_16x16x32_f16(false, a_a[kc], false, b_l,
                                                       (short)0, c, false, false);
            const v16h b_r = load_frag32(Wr + (size_t)(n0 + idx) * 64 + 32 * kc, half);
            c = __builtin_amdgcn_wmma_f32_16x16x32_f16(false, a_h[kc], false, b_r,
                                                       (short)0, c, false, false);
        }
        const float bias = bl[n0 + idx];
#pragma unroll
        for (int r = 0; r < 8; ++r)
            out[(size_t)(m0 + r + 8 * half) * 32 + n0 + idx] = c[r] + bias;
    }
}

// ---------------------------------------------------------------------------
// Host-side launch: zero -> scatter1 -> gemm1 -> scatter2 -> gemm2
// Workspace layout (floats): agg1[n*32] | h[n*64] | agg2[n*64]  (= 64 MB)
// ---------------------------------------------------------------------------
extern "C" void kernel_launch(void* const* d_in, const int* in_sizes, int n_in,
                              void* d_out, int out_size, void* d_ws, size_t ws_size,
                              hipStream_t stream) {
    const float* x     = (const float*)d_in[0];
    const int*   edges = (const int*)d_in[1];     // [0..E)=src, [E..2E)=dst
    const float* Wl1   = (const float*)d_in[2];   // [64,32]
    const float* bl1   = (const float*)d_in[3];   // [64]
    const float* Wr1   = (const float*)d_in[4];   // [64,32]
    const float* Wl2   = (const float*)d_in[5];   // [32,64]
    const float* bl2   = (const float*)d_in[6];   // [32]
    const float* Wr2   = (const float*)d_in[7];   // [32,64]
    float*       out   = (float*)d_out;

    const int n  = in_sizes[0] / 32;              // 100000 nodes
    const int nE = in_sizes[1] / 2;               // 1600000 edges

    float* agg1 = (float*)d_ws;
    float* h    = agg1 + (size_t)n * 32;
    float* agg2 = h    + (size_t)n * 64;

    // 1) zero the atomic accumulators
    {
        long long c1 = (long long)n * 32 / 4;
        zero4_kernel<<<(int)((c1 + 255) / 256), 256, 0, stream>>>((float4*)agg1, c1);
        long long c2 = (long long)n * 64 / 4;
        zero4_kernel<<<(int)((c2 + 255) / 256), 256, 0, stream>>>((float4*)agg2, c2);
    }

    // 2) agg1 += scatter of x (d=32, 8 float4 chunks/edge)
    {
        long long t = (long long)nE * 8;
        scatter_kernel<<<(int)((t + 255) / 256), 256, 0, stream>>>(
            edges, edges + nE, x, agg1, nE, 3);
    }

    const int nTiles = n / 16;                    // 6250, exact
    const int gemmBlocks = (nTiles + 7) / 8;      // 8 waves / 256-thread block

    // 3) h = relu(agg1 @ Wl1^T + b1 + x @ Wr1^T)
    gemm1_kernel<<<gemmBlocks, 256, 0, stream>>>(agg1, x, Wl1, bl1, Wr1, h, nTiles);

    // 4) agg2 += scatter of h (d=64, 16 float4 chunks/edge)
    {
        long long t = (long long)nE * 16;
        scatter_kernel<<<(int)((t + 255) / 256), 256, 0, stream>>>(
            edges, edges + nE, h, agg2, nE, 4);
    }

    // 5) out = agg2 @ Wl2^T + b2 + h @ Wr2^T
    gemm2_kernel<<<gemmBlocks, 256, 0, stream>>>(agg2, h, Wl2, bl2, Wr2, out, nTiles);
}
